// CoreContextAwareAttention_18184891532020
// MI455X (gfx1250) — compile-verified, matrix operand order checked by reading, not run on hardware
//
#include <hip/hip_runtime.h>
#include <hip/hip_bf16.h>
#include <cstdint>

// ---------------------------------------------------------------------------
// Types for CDNA5 WMMA
// ---------------------------------------------------------------------------
typedef __attribute__((ext_vector_type(2)))  float  v2f;
typedef __attribute__((ext_vector_type(8)))  float  v8f;
typedef __attribute__((ext_vector_type(16))) __bf16 v16bf;

#define BQ 8
#define SQ 8192
#define DQ 1024
#define GQ 512          // groups per batch
#define GSQ 16          // group size
#define KSEL 64         // top-k groups
#define HQ 16           // heads
#define HDQ 64          // head dim
#define FQ 256          // DH4 (scoring hidden)

// ---------------------------------------------------------------------------
// Weight transpose + fp32 -> bf16 (WT[n][d] = W[d][n]), 1024x1024
// ---------------------------------------------------------------------------
__global__ void k_transpose_bf16(const float* __restrict__ W, __bf16* __restrict__ WT) {
  __shared__ float tile[32][33];
  int tx = threadIdx.x;
  int x  = blockIdx.x * 32 + tx;
  int y0 = blockIdx.y * 32;
  for (int i = threadIdx.y; i < 32; i += 8)
    tile[i][tx] = W[(size_t)(y0 + i) * DQ + x];
  __syncthreads();
  int xo = blockIdx.y * 32 + tx;
  int y1 = blockIdx.x * 32;
  for (int i = threadIdx.y; i < 32; i += 8)
    WT[(size_t)(y1 + i) * DQ + xo] = (__bf16)tile[tx][i];
}

// ---------------------------------------------------------------------------
// Group mean-pool: hidden [B,S,D] -> groups [B*G, D]   (256 MB streamed read)
// ---------------------------------------------------------------------------
__global__ void k_meanpool(const float* __restrict__ hs, float* __restrict__ G) {
  size_t t  = (size_t)blockIdx.x * 256 + threadIdx.x;   // B*G*D threads
  size_t bg = t >> 10;
  int d     = (int)(t & 1023);
  const float* p = hs + bg * (size_t)(GSQ * DQ) + d;
  float s = 0.f;
  #pragma unroll
  for (int i = 0; i < GSQ; ++i) s += p[(size_t)i * DQ];
  G[t] = s * (1.f / GSQ);
}

// ---------------------------------------------------------------------------
// Scoring MLP with fp32 WMMA (ranking-sensitive -> keep fp32):
// scores[bg] = Ws2 . relu(groups[bg,:] @ Ws1 + bs1)      (softmax/bs2 skipped)
// 4 waves / block; each wave owns 4 n-tiles (32 acc VGPRs -> no spills);
// A tile (16 groups x 64 k) staged in LDS once per k-chunk, shared by waves.
// Final score combine through LDS in fixed wave order => deterministic.
// ---------------------------------------------------------------------------
__global__ void k_score(const float* __restrict__ G, const float* __restrict__ Ws1,
                        const float* __restrict__ bs1, const float* __restrict__ Ws2,
                        float* __restrict__ scores) {
  __shared__ float At[16][68];                    // 16 rows x 64 k, padded
  __shared__ float spbuf[4][16];                  // per-wave partial scores
  int m0 = blockIdx.x * 16;
  int wv = threadIdx.x >> 5;                      // wave 0..3 -> n-tiles wv*4..wv*4+3
  int l = threadIdx.x & 31, r = l & 15, h = l >> 4;

  v8f acc[4];
  #pragma unroll
  for (int i = 0; i < 4; ++i) acc[i] = (v8f){0,0,0,0,0,0,0,0};

  for (int k0 = 0; k0 < DQ; k0 += 64) {
    __syncthreads();
    for (int idx = threadIdx.x; idx < 16 * 16; idx += 128) {  // 16 rows x 16 quads
      int rr = idx >> 4, c4 = idx & 15;
      float4 v = *(const float4*)(G + (size_t)(m0 + rr) * DQ + k0 + c4 * 4);
      *(float4*)&At[rr][c4 * 4] = v;
    }
    __syncthreads();
    for (int ks = 0; ks < 64; ks += 4) {
      int kb = ks + 2 * h;                        // per-half K pair base
      v2f a; a.x = At[r][kb]; a.y = At[r][kb + 1];
      const float* B0 = Ws1 + (size_t)(k0 + kb) * FQ + wv * 64 + r;
      #pragma unroll
      for (int ntl = 0; ntl < 4; ++ntl) {
        v2f bv; bv.x = B0[ntl * 16]; bv.y = B0[FQ + ntl * 16];
        acc[ntl] = __builtin_amdgcn_wmma_f32_16x16x4_f32(
            false, a, false, bv, (short)0, acc[ntl], false, false);
      }
    }
  }
  // epilogue: relu + dot with Ws2, reduce over n within each 16-lane half
  float sp[8];
  #pragma unroll
  for (int j = 0; j < 8; ++j) sp[j] = 0.f;
  #pragma unroll
  for (int ntl = 0; ntl < 4; ++ntl) {
    int n = (wv * 4 + ntl) * 16 + r;
    float b1 = bs1[n], w2 = Ws2[n];
    #pragma unroll
    for (int j = 0; j < 8; ++j) {
      float hv = acc[ntl][j] + b1;
      hv = hv > 0.f ? hv : 0.f;
      sp[j] += hv * w2;
    }
  }
  #pragma unroll
  for (int j = 0; j < 8; ++j) {
    float v = sp[j];
    v += __shfl_xor(v, 1, 16);
    v += __shfl_xor(v, 2, 16);
    v += __shfl_xor(v, 4, 16);
    v += __shfl_xor(v, 8, 16);
    sp[j] = v;
  }
  if (r == 0) {
    #pragma unroll
    for (int j = 0; j < 8; ++j) spbuf[wv][h * 8 + j] = sp[j];
  }
  __syncthreads();
  if (threadIdx.x < 16) {
    int row = threadIdx.x;
    scores[m0 + row] = (spbuf[0][row] + spbuf[1][row])
                     + (spbuf[2][row] + spbuf[3][row]);
  }
}

// ---------------------------------------------------------------------------
// Top-K by rank counting (matches jax top_k lowest-index tie-break).
// One 512-thread block per batch.
// ---------------------------------------------------------------------------
__global__ void k_topk(const float* __restrict__ scores, int* __restrict__ selidx,
                       int* __restrict__ slot) {
  __shared__ float s[GQ];
  int b = blockIdx.x, g = threadIdx.x;
  float v = scores[b * GQ + g];
  s[g] = v;
  __syncthreads();
  int rank = 0;
  for (int j = 0; j < GQ; ++j) {
    float sj = s[j];
    rank += (sj > v || (sj == v && j < g)) ? 1 : 0;
  }
  if (rank < KSEL) { selidx[b * KSEL + rank] = g; slot[b * GQ + g] = rank; }
  else             { slot[b * GQ + g] = -1; }
}

// ---------------------------------------------------------------------------
// Gather selected group reps, convert to bf16: sel [B*K, D]
// ---------------------------------------------------------------------------
__global__ void k_gather(const float* __restrict__ G, const int* __restrict__ selidx,
                         __bf16* __restrict__ sel) {
  size_t t  = (size_t)blockIdx.x * 256 + threadIdx.x;   // B*K*D
  size_t bk = t >> 10;
  int d = (int)(t & 1023);
  int b = (int)(bk >> 6), k = (int)(bk & 63);
  int g = selidx[b * KSEL + k];
  sel[t] = (__bf16)G[((size_t)(b * GQ + g) << 10) + d];
}

// ---------------------------------------------------------------------------
// bf16 WMMA operand loader: per-lane row m0+l%16, two contiguous 16B chunks
// at k = k0 + 8*(l/16) and +16  -> matches 16-bit A/B VGPR layout exactly.
// ---------------------------------------------------------------------------
__device__ __forceinline__ v16bf load_tile16(const __bf16* p, int row0, int k0, int ld) {
  int l = threadIdx.x & 31;
  int r = l & 15, h = l >> 4;
  const __bf16* base = p + (size_t)(row0 + r) * ld + k0 + h * 8;
  union { uint4 u[2]; v16bf v; } c;
  c.u[0] = *(const uint4*)(base);
  c.u[1] = *(const uint4*)(base + 16);
  return c.v;
}

// ---------------------------------------------------------------------------
// Generic GEMM, 32x32 tile per wave (4 WMMAs per 2 A + 2 B loads):
// out = A[M,K] @ Bt[N,K]^T (Bt pre-transposed).
// mode 0: bf16 row-major out   mode 1: V^T per-head layout   mode 2: f32 out
// ---------------------------------------------------------------------------
__global__ void k_gemm_bf16(const __bf16* __restrict__ A, const __bf16* __restrict__ Bt,
                            void* __restrict__ out, int Kdim, int N, int mode) {
  int m0 = blockIdx.x * 32;
  int n0 = blockIdx.y * 32;
  int l = threadIdx.x & 31, r = l & 15, h = l >> 4;
  v8f acc[2][2];
  #pragma unroll
  for (int i = 0; i < 2; ++i)
    #pragma unroll
    for (int j = 0; j < 2; ++j) acc[i][j] = (v8f){0,0,0,0,0,0,0,0};

  for (int k0 = 0; k0 < Kdim; k0 += 32) {
    if (k0 + 128 < Kdim) {   // global_prefetch_b8 a few tiles ahead
      __builtin_prefetch(A  + (size_t)(m0 + r) * Kdim + k0 + 128, 0, 1);
      __builtin_prefetch(Bt + (size_t)(n0 + r) * Kdim + k0 + 128, 0, 1);
    }
    v16bf a0 = load_tile16(A,  m0,      k0, Kdim);
    v16bf a1 = load_tile16(A,  m0 + 16, k0, Kdim);
    v16bf b0 = load_tile16(Bt, n0,      k0, Kdim);
    v16bf b1 = load_tile16(Bt, n0 + 16, k0, Kdim);
    acc[0][0] = __builtin_amdgcn_wmma_f32_16x16x32_bf16(false, a0, false, b0, (short)0, acc[0][0], false, false);
    acc[0][1] = __builtin_amdgcn_wmma_f32_16x16x32_bf16(false, a0, false, b1, (short)0, acc[0][1], false, false);
    acc[1][0] = __builtin_amdgcn_wmma_f32_16x16x32_bf16(false, a1, false, b0, (short)0, acc[1][0], false, false);
    acc[1][1] = __builtin_amdgcn_wmma_f32_16x16x32_bf16(false, a1, false, b1, (short)0, acc[1][1], false, false);
  }
  #pragma unroll
  for (int i = 0; i < 2; ++i)
    #pragma unroll
    for (int jj = 0; jj < 2; ++jj) {
      int ms = m0 + i * 16, ns = n0 + jj * 16;
      if (mode == 0) {
        __bf16* o = (__bf16*)out;
        #pragma unroll
        for (int j = 0; j < 8; ++j)
          o[(size_t)(ms + h * 8 + j) * N + ns + r] = (__bf16)acc[i][jj][j];
      } else if (mode == 1) {  // V^T: o[((b*H + hh)*HD + dh)*K + key]
        __bf16* o = (__bf16*)out;
        #pragma unroll
        for (int j = 0; j < 8; ++j) {
          int m = ms + h * 8 + j, n = ns + r;
          int b_ = m >> 6, key = m & 63, hh = n >> 6, dh = n & 63;
          o[(size_t)(((b_ * HQ + hh) * HDQ + dh) * KSEL) + key] = (__bf16)acc[i][jj][j];
        }
      } else {
        float* o = (float*)out;
        #pragma unroll
        for (int j = 0; j < 8; ++j)
          o[(size_t)(ms + h * 8 + j) * N + ns + r] = acc[i][jj][j];
      }
    }
}

// ---------------------------------------------------------------------------
// Attention per (b, head): S = Q K^T / 8, softmax rows, ctx = P V. One wave.
// ---------------------------------------------------------------------------
__global__ void k_attn(const __bf16* __restrict__ Qb, const __bf16* __restrict__ Kb,
                       const __bf16* __restrict__ Vt, __bf16* __restrict__ Ctx) {
  __shared__ float  Sf[64][64];
  __shared__ __bf16 P[64][64];
  int bh = blockIdx.x;
  int b = bh >> 4, hh = bh & 15;
  int l = threadIdx.x & 31, r = l & 15, h = l >> 4;

  const __bf16* Qp = Qb + (size_t)(b * KSEL) * DQ + hh * HDQ;
  const __bf16* Kp = Kb + (size_t)(b * KSEL) * DQ + hh * HDQ;

  #pragma unroll
  for (int mt = 0; mt < 4; ++mt) {
    v16bf aq0 = load_tile16(Qp, mt * 16, 0,  DQ);
    v16bf aq1 = load_tile16(Qp, mt * 16, 32, DQ);
    #pragma unroll
    for (int nt = 0; nt < 4; ++nt) {
      v8f acc = (v8f){0,0,0,0,0,0,0,0};
      v16bf b0 = load_tile16(Kp, nt * 16, 0,  DQ);   // B[k][n]=K[n][k]: QK^T
      v16bf b1 = load_tile16(Kp, nt * 16, 32, DQ);
      acc = __builtin_amdgcn_wmma_f32_16x16x32_bf16(false, aq0, false, b0, (short)0, acc, false, false);
      acc = __builtin_amdgcn_wmma_f32_16x16x32_bf16(false, aq1, false, b1, (short)0, acc, false, false);
      #pragma unroll
      for (int j = 0; j < 8; ++j)
        Sf[mt * 16 + h * 8 + j][nt * 16 + r] = acc[j] * 0.125f;
    }
  }
  __syncthreads();

  for (int row = l; row < 64; row += 32) {
    float m = -1e30f;
    for (int c = 0; c < 64; ++c) m = fmaxf(m, Sf[row][c]);
    float sum = 0.f;
    for (int c = 0; c < 64; ++c) sum += __expf(Sf[row][c] - m);
    float inv = 1.f / sum;
    for (int c = 0; c < 64; ++c) P[row][c] = (__bf16)(__expf(Sf[row][c] - m) * inv);
  }
  __syncthreads();

  const __bf16* Vp = Vt + (size_t)(b * HQ + hh) * (HDQ * KSEL);  // [dh][key]
  #pragma unroll
  for (int mt = 0; mt < 4; ++mt) {
    v16bf ap0 = load_tile16(&P[0][0], mt * 16, 0,  64);
    v16bf ap1 = load_tile16(&P[0][0], mt * 16, 32, 64);
    #pragma unroll
    for (int nt = 0; nt < 4; ++nt) {
      v8f acc = (v8f){0,0,0,0,0,0,0,0};
      v16bf b0 = load_tile16(Vp, nt * 16, 0,  KSEL);  // B[k][dh]=Vt[dh][k]
      v16bf b1 = load_tile16(Vp, nt * 16, 32, KSEL);
      acc = __builtin_amdgcn_wmma_f32_16x16x32_bf16(false, ap0, false, b0, (short)0, acc, false, false);
      acc = __builtin_amdgcn_wmma_f32_16x16x32_bf16(false, ap1, false, b1, (short)0, acc, false, false);
      #pragma unroll
      for (int j = 0; j < 8; ++j)
        Ctx[(size_t)(b * KSEL + mt * 16 + h * 8 + j) * DQ + hh * HDQ + nt * 16 + r]
            = (__bf16)acc[j];
    }
  }
}

// ---------------------------------------------------------------------------
// Fused zero-fill + scatter-broadcast (256 MB write, float4 coalesced)
// ---------------------------------------------------------------------------
__global__ void k_scatter(const float4* __restrict__ outg, const int* __restrict__ slot,
                          float4* __restrict__ out) {
  size_t t = (size_t)blockIdx.x * 256 + threadIdx.x;   // B*S*D/4 quads
  int d4 = (int)(t & 255);
  size_t row = t >> 8;
  size_t b = row >> 13;
  int s = (int)(row & 8191);
  int g = s >> 4;
  int sl = slot[b * GQ + g];
  float4 v = make_float4(0.f, 0.f, 0.f, 0.f);
  if (sl >= 0) v = outg[(((size_t)b * KSEL + sl) << 8) + d4];
  out[t] = v;
}

// ---------------------------------------------------------------------------
// Launch
// ---------------------------------------------------------------------------
extern "C" void kernel_launch(void* const* d_in, const int* in_sizes, int n_in,
                              void* d_out, int out_size, void* d_ws, size_t ws_size,
                              hipStream_t stream) {
  (void)in_sizes; (void)n_in; (void)out_size; (void)ws_size;
  const float* hs  = (const float*)d_in[0];
  const float* Wq  = (const float*)d_in[1];
  const float* Wk  = (const float*)d_in[2];
  const float* Wv  = (const float*)d_in[3];
  const float* Wo  = (const float*)d_in[4];
  const float* Ws1 = (const float*)d_in[5];
  const float* bs1 = (const float*)d_in[6];
  const float* Ws2 = (const float*)d_in[7];
  // bs2 (d_in[8]) shifts all scores equally -> irrelevant to top-k; softmax skipped.

  char* ws = (char*)d_ws;                                   // ~32 MB used
  float*  G      = (float*)(ws);                            // 16 MB  groups f32
  __bf16* WqT    = (__bf16*)(ws + (16u << 20));             // 4 x 2 MB bf16 W^T
  __bf16* WkT    = WqT + (1u << 20);
  __bf16* WvT    = WkT + (1u << 20);
  __bf16* WoT    = WvT + (1u << 20);
  __bf16* sel    = (__bf16*)(ws + (24u << 20));             // 1 MB
  __bf16* Qb     = (__bf16*)(ws + (25u << 20));             // 1 MB
  __bf16* Kb     = (__bf16*)(ws + (26u << 20));             // 1 MB
  __bf16* Vt     = (__bf16*)(ws + (27u << 20));             // 1 MB
  __bf16* Ctx    = (__bf16*)(ws + (28u << 20));             // 1 MB
  float*  outg   = (float*)(ws + (29u << 20));              // 2 MB
  float*  scores = (float*)(ws + (31u << 20));              // 16 KB
  int*    selidx = (int*)(ws + (31u << 20) + (64u << 10));  // 2 KB
  int*    slot   = (int*)(ws + (31u << 20) + (128u << 10)); // 16 KB

  dim3 tb(32, 8);
  k_transpose_bf16<<<dim3(32, 32), tb, 0, stream>>>(Wq, WqT);
  k_transpose_bf16<<<dim3(32, 32), tb, 0, stream>>>(Wk, WkT);
  k_transpose_bf16<<<dim3(32, 32), tb, 0, stream>>>(Wv, WvT);
  k_transpose_bf16<<<dim3(32, 32), tb, 0, stream>>>(Wo, WoT);

  k_meanpool<<<(BQ * GQ * DQ) / 256, 256, 0, stream>>>(hs, G);
  k_score<<<(BQ * GQ) / 16, 128, 0, stream>>>(G, Ws1, bs1, Ws2, scores);
  k_topk<<<BQ, GQ, 0, stream>>>(scores, selidx, slot);
  k_gather<<<(BQ * KSEL * DQ) / 256, 256, 0, stream>>>(G, selidx, sel);

  dim3 gg(BQ * KSEL / 32, DQ / 32);
  k_gemm_bf16<<<gg, 32, 0, stream>>>(sel, WqT, Qb, DQ, DQ, 0);
  k_gemm_bf16<<<gg, 32, 0, stream>>>(sel, WkT, Kb, DQ, DQ, 0);
  k_gemm_bf16<<<gg, 32, 0, stream>>>(sel, WvT, Vt, DQ, DQ, 1);

  k_attn<<<BQ * HQ, 32, 0, stream>>>(Qb, Kb, Vt, Ctx);

  k_gemm_bf16<<<gg, 32, 0, stream>>>(Ctx, WoT, outg, DQ, DQ, 2);

  k_scatter<<<(BQ * SQ * DQ / 4) / 256, 256, 0, stream>>>(
      (const float4*)outg, slot, (float4*)d_out);
}
